// ESAMM_4655744549593
// MI455X (gfx1250) — matrix operand order repaired and checked by reading, not compiled, not run on hardware
//
#include <hip/hip_runtime.h>
#include <hip/hip_bf16.h>

typedef __attribute__((ext_vector_type(16))) _Float16 v16h;
typedef __attribute__((ext_vector_type(8)))  _Float16 v8h;
typedef __attribute__((ext_vector_type(8)))  float    v8f;

#define NB   8
#define CIN  64
#define HH   256
#define WW   256
#define COUT 128
#define KCH  32            // key/value channels == heads
#define HWSZ 65536

__device__ inline v8h zero8h() {
  v8h r;
  for (int i = 0; i < 8; ++i) r[i] = (_Float16)0.f;
  return r;
}

// ---------------------------------------------------------------------------
// K0a: repack conv weights (f32 OIHW) into WMMA A-fragment order, f16.
// K order: k = tap*64 + ci  (tap = ky*3+kx). Layout [mtile=8][ktile=18][lane=32][16]
// lane<16 : K=j
// lane>=16: K=j+16
// ---------------------------------------------------------------------------
__global__ void k_pack_conv(const float* __restrict__ w, _Float16* __restrict__ apack) {
  int idx = blockIdx.x * 256 + threadIdx.x;
  if (idx >= 8 * 18 * 32 * 16) return;
  int j    = idx & 15;
  int lane = (idx >> 4) & 31;
  int kt   = (idx >> 9) % 18;
  int mt   = idx / (16 * 32 * 18);
  int klo  = ((lane >> 4) << 4) + j;
  int k    = kt * 32 + klo;
  int tap  = k >> 6, ci = k & 63;
  int M    = mt * 16 + (lane & 15);
  apack[idx] = (_Float16)w[(M * 64 + ci) * 9 + tap];
}

// K0b: pack keys_w (rows 0..31) + values_w (rows 32..63) the same way.
// [mtile=4][ktile=4][lane=32][16], K = channel c in [0,128)
__global__ void k_pack_proj(const float* __restrict__ kw, const float* __restrict__ vw,
                            _Float16* __restrict__ ppack) {
  int idx = blockIdx.x * 256 + threadIdx.x;
  if (idx >= 4 * 4 * 32 * 16) return;
  int j    = idx & 15;
  int lane = (idx >> 4) & 31;
  int kt   = (idx >> 9) & 3;
  int mt   = idx >> 11;
  int klo  = ((lane >> 4) << 4) + j;
  int c    = kt * 32 + klo;
  int row  = mt * 16 + (lane & 15);
  float v  = (row < 32) ? kw[row * 128 + c] : vw[(row - 32) * 128 + c];
  ppack[idx] = (_Float16)v;
}

// ---------------------------------------------------------------------------
// K1: mixed = (x-y)+x*y, written f16 in pixel-major layout [n][h][w][ci]
//     (so the conv kernel's LDS B-fragments are contiguous channel runs).
//     LDS transpose for coalesced stores.
// ---------------------------------------------------------------------------
__global__ void __launch_bounds__(256) k_mix(const float* __restrict__ x,
                                             const float* __restrict__ y,
                                             _Float16* __restrict__ mixedT) {
  __shared__ _Float16 t[64 * 65];
  int n = blockIdx.z, h = blockIdx.y, w0 = blockIdx.x * 64;
  int tid = threadIdx.x;
  #pragma unroll 4
  for (int pass = 0; pass < 16; ++pass) {
    int e  = pass * 256 + tid;            // 4096 elements: 64 ci x 64 px
    int ci = e >> 6, px = e & 63;
    size_t g = ((size_t)(n * 64 + ci) * 256 + h) * 256 + w0 + px;
    float xv = x[g], yv = y[g];
    t[ci * 65 + px] = (_Float16)((xv - yv) + xv * yv);
  }
  __syncthreads();
  int px = tid >> 2, q = tid & 3;         // 4 threads per pixel, 16 ch each
  v8h v0 = zero8h(), v1 = zero8h();
  #pragma unroll
  for (int i = 0; i < 8; ++i) {
    v0[i] = t[(q * 16 + i) * 65 + px];
    v1[i] = t[(q * 16 + 8 + i) * 65 + px];
  }
  size_t o = ((size_t)((n * 256 + h) * 256) + w0 + px) * 64 + q * 16;
  *(v8h*)(mixedT + o)     = v0;
  *(v8h*)(mixedT + o + 8) = v1;
}

// ---------------------------------------------------------------------------
// K2: 3x3 conv as implicit GEMM with v_wmma_f32_16x16x32_f16.
// Block = 256 thr (8 waves), one (n, h-row, 64-px strip), all 128 c_out.
// Wave w owns mtile=w (16 c_out) x 64 px (4 accumulators), K loop = 18.
// Epilogue: +bias, PReLU, store z to d_out, fused InstanceNorm partial stats.
// ---------------------------------------------------------------------------
__global__ void __launch_bounds__(256, 2)
k_conv(const _Float16* __restrict__ mixedT, const _Float16* __restrict__ apack,
       const float* __restrict__ conv_b, const float* __restrict__ prelu_a,
       float* __restrict__ zout, float* __restrict__ ssum, float* __restrict__ ssq) {
  __shared__ _Float16 lin[3 * 66 * 72];   // [3 rows][66 cols][64 ci pad 72]
  int n = blockIdx.z, h = blockIdx.y, w0 = blockIdx.x * 64;
  int tid = threadIdx.x;

  for (int task = tid; task < 3 * 66 * 4; task += 256) {
    int q = task & 3, pc = task >> 2;
    int row = pc / 66, col = pc % 66;
    int hh = h + row - 1, ww = w0 + col - 1;
    v8h a = zero8h(), b = zero8h();
    if (hh >= 0 && hh < HH && ww >= 0 && ww < WW) {
      const _Float16* src = mixedT + ((size_t)((n * 256 + hh) * 256 + ww)) * 64 + q * 16;
      a = *(const v8h*)src;
      b = *(const v8h*)(src + 8);
    }
    _Float16* dst = lin + (row * 66 + col) * 72 + q * 16;
    *(v8h*)dst = a;
    *(v8h*)(dst + 8) = b;
  }
  __syncthreads();

  int wave = tid >> 5, lane = tid & 31;
  int mtile = wave;
  int lo = lane & 15, hiHalf = (lane >> 4) << 4;

  v8f acc[4];
  #pragma unroll
  for (int i = 0; i < 4; ++i)
    for (int j = 0; j < 8; ++j) acc[i][j] = 0.f;

  const _Float16* abase = apack + (size_t)mtile * 18 * 32 * 16;
  for (int kt = 0; kt < 18; ++kt) {
    __builtin_prefetch(abase + ((kt + 1) * 32 + lane) * 16, 0, 0);
    v16h afrag = *(const v16h*)(abase + (kt * 32 + lane) * 16);
    int tap = kt >> 1;
    int cib = ((kt & 1) << 5) + hiHalf;
    int ky = tap / 3, kx = tap % 3;
    const _Float16* bp = lin + (ky * 66 + lo + kx) * 72 + cib;
    #pragma unroll
    for (int nt = 0; nt < 4; ++nt) {
      const _Float16* p = bp + nt * 16 * 72;
      v8h blo = *(const v8h*)p;
      v8h bhi = *(const v8h*)(p + 8);
      v16h bfrag = __builtin_shufflevector(blo, bhi, 0,1,2,3,4,5,6,7,8,9,10,11,12,13,14,15);
      acc[nt] = __builtin_amdgcn_wmma_f32_16x16x32_f16(false, afrag, false, bfrag,
                                                       (short)0, acc[nt], false, false);
    }
  }

  float pa = prelu_a[0];
  #pragma unroll
  for (int r = 0; r < 8; ++r) {
    int c = mtile * 16 + r + ((lane >> 4) << 3);
    float bias = conv_b[c];
    float s = 0.f, q = 0.f;
    size_t obase = (((size_t)(n * 128 + c)) << 16) + (size_t)h * 256 + w0 + lo;
    #pragma unroll
    for (int nt = 0; nt < 4; ++nt) {
      float v = acc[nt][r] + bias;
      v = (v >= 0.f) ? v : pa * v;       // PReLU
      zout[obase + nt * 16] = v;
      s += v; q += v * v;
    }
    #pragma unroll
    for (int m = 1; m <= 8; m <<= 1) {   // reduce 16-lane half (channels differ per half)
      s += __shfl_xor(s, m, 32);
      q += __shfl_xor(q, m, 32);
    }
    if (lo == 0) {
      atomicAdd(&ssum[n * 128 + c], s);
      atomicAdd(&ssq [n * 128 + c], q);
    }
  }
}

// K3: finalize instance-norm stats
__global__ void k_finstats(const float* __restrict__ ssum, const float* __restrict__ ssq,
                           float* __restrict__ mu, float* __restrict__ rsig) {
  int i = blockIdx.x * 256 + threadIdx.x;
  if (i < NB * COUT) {
    float m = ssum[i] * (1.f / 65536.f);
    float v = ssq[i] * (1.f / 65536.f) - m * m;
    mu[i] = m;
    rsig[i] = rsqrtf(v + 1e-5f);
  }
}

// ---------------------------------------------------------------------------
// K4: keys/values projections via WMMA on normalized z tile (64 px x 128 c),
// then exp(keys) and fused partial Σexp, Σexp*val per (n,head) via atomics.
// (queries softmax over singleton axis == 1, so attention reduces to this.)
// ---------------------------------------------------------------------------
__global__ void __launch_bounds__(256, 2)
k_attn(const float* __restrict__ zout, const _Float16* __restrict__ ppack,
       const float* __restrict__ mu, const float* __restrict__ rsig,
       const float* __restrict__ keys_b, const float* __restrict__ values_b,
       float* __restrict__ Sacc, float* __restrict__ Cacc) {
  __shared__ _Float16 zn[64 * 136];       // [px][128 c pad 136]
  __shared__ float dval[64 * 65];         // rows 0..31 exp(keys), 32..63 values
  int n = blockIdx.y;
  int p0 = blockIdx.x * 64;
  int tid = threadIdx.x;

  for (int task = tid; task < 512; task += 256) {
    int c = task >> 2, seg = task & 3;
    float m = mu[n * 128 + c], rs = rsig[n * 128 + c];
    const float* src = zout + (((size_t)(n * 128 + c)) << 16) + p0 + seg * 16;
    #pragma unroll
    for (int i = 0; i < 16; ++i)
      zn[(seg * 16 + i) * 136 + c] = (_Float16)((src[i] - m) * rs);
  }
  __syncthreads();

  int wave = tid >> 5, lane = tid & 31;
  int mtile = wave >> 1;
  int ntb = (wave & 1) * 2;
  int lo = lane & 15, hiHalf = (lane >> 4) << 4;

  v8f acc[2];
  #pragma unroll
  for (int i = 0; i < 2; ++i)
    for (int j = 0; j < 8; ++j) acc[i][j] = 0.f;

  #pragma unroll
  for (int kt = 0; kt < 4; ++kt) {
    v16h afrag = *(const v16h*)(ppack + ((mtile * 4 + kt) * 32 + lane) * 16);
    int cb = kt * 32 + hiHalf;
    #pragma unroll
    for (int t = 0; t < 2; ++t) {
      const _Float16* p = zn + ((ntb + t) * 16 + lo) * 136 + cb;
      v8h blo = *(const v8h*)p;
      v8h bhi = *(const v8h*)(p + 8);
      v16h bfrag = __builtin_shufflevector(blo, bhi, 0,1,2,3,4,5,6,7,8,9,10,11,12,13,14,15);
      acc[t] = __builtin_amdgcn_wmma_f32_16x16x32_f16(false, afrag, false, bfrag,
                                                      (short)0, acc[t], false, false);
    }
  }

  #pragma unroll
  for (int r = 0; r < 8; ++r) {
    int row = mtile * 16 + r + ((lane >> 4) << 3);
    float bias = (row < 32) ? keys_b[row] : values_b[row - 32];
    #pragma unroll
    for (int t = 0; t < 2; ++t) {
      int px = (ntb + t) * 16 + lo;
      float v = acc[t][r] + bias;
      dval[row * 65 + px] = (row < 32) ? __expf(v) : v;   // |kr| ~ 5 max: safe w/o max-sub
    }
  }
  __syncthreads();

  int j = tid & 31, seg = tid >> 5;
  float S = 0.f, C = 0.f;
  #pragma unroll
  for (int i = 0; i < 8; ++i) {
    int px = seg * 8 + i;
    float e = dval[j * 65 + px];
    float v = dval[(j + 32) * 65 + px];
    S += e;
    C += e * v;
  }
  atomicAdd(&Sacc[n * 32 + j], S);
  atomicAdd(&Cacc[n * 32 + j], C);
}

// K5: context = C/S (per n,head); reproj_vec[n][c] = reproj_w @ context + b
__global__ void k_reproj(const float* __restrict__ Sacc, const float* __restrict__ Cacc,
                         const float* __restrict__ rw, const float* __restrict__ rb,
                         float* __restrict__ rvec) {
  int n = blockIdx.x, c = threadIdx.x;
  float a = rb[c];
  #pragma unroll
  for (int j = 0; j < 32; ++j)
    a += rw[c * 32 + j] * (Cacc[n * 32 + j] / Sacc[n * 32 + j]);
  rvec[n * 128 + c] = a;
}

// K6: out = (z - mu)*rsig + rvec  (in place on d_out, float4)
__global__ void __launch_bounds__(256) k_final(float* __restrict__ out,
                                               const float* __restrict__ mu,
                                               const float* __restrict__ rsig,
                                               const float* __restrict__ rvec) {
  size_t i4 = (size_t)blockIdx.x * 256 + threadIdx.x;   // 16384 float4 per (n,c) plane
  int plane = (int)(i4 >> 14);
  float m = mu[plane], rs = rsig[plane], rv = rvec[plane];
  float4 v = ((float4*)out)[i4];
  v.x = (v.x - m) * rs + rv;
  v.y = (v.y - m) * rs + rv;
  v.z = (v.z - m) * rs + rv;
  v.w = (v.w - m) * rs + rv;
  ((float4*)out)[i4] = v;
}

extern "C" void kernel_launch(void* const* d_in, const int* in_sizes, int n_in,
                              void* d_out, int out_size, void* d_ws, size_t ws_size,
                              hipStream_t stream) {
  (void)in_sizes; (void)n_in; (void)out_size; (void)ws_size;
  const float* x        = (const float*)d_in[0];
  const float* y        = (const float*)d_in[1];
  const float* conv_w   = (const float*)d_in[2];
  const float* conv_b   = (const float*)d_in[3];
  const float* prelu_a  = (const float*)d_in[4];
  const float* keys_w   = (const float*)d_in[5];
  const float* keys_b   = (const float*)d_in[6];
  // d_in[7]/d_in[8] (queries): softmax over size-1 axis == 1 -> drop out of the math
  const float* values_w = (const float*)d_in[9];
  const float* values_b = (const float*)d_in[10];
  const float* reproj_w = (const float*)d_in[11];
  const float* reproj_b = (const float*)d_in[12];
  float* out = (float*)d_out;          // doubles as z (conv output) scratch

  char* w = (char*)d_ws;
  _Float16* mixedT = (_Float16*)(w);                       // 67,108,864 B
  _Float16* apack  = (_Float16*)(w + 67108864);            //    147,456 B
  _Float16* ppack  = (_Float16*)(w + 67256320);            //     16,384 B
  float*    ssum   = (float*)(w + 67272704);               //      4,096 B
  float*    ssq    = (float*)(w + 67276800);               //      4,096 B
  float*    muv    = (float*)(w + 67280896);               //      4,096 B
  float*    rsg    = (float*)(w + 67284992);               //      4,096 B
  float*    Sacc   = (float*)(w + 67289088);               //      1,024 B
  float*    Cacc   = (float*)(w + 67290112);               //      1,024 B
  float*    rvec   = (float*)(w + 67291136);               //      4,096 B

  (void)hipMemsetAsync(ssum, 0, 8192, stream);    // ssum + ssq
  (void)hipMemsetAsync(Sacc, 0, 2048, stream);    // Sacc + Cacc

  k_pack_conv<<<288, 256, 0, stream>>>(conv_w, apack);
  k_pack_proj<<<32, 256, 0, stream>>>(keys_w, values_w, ppack);
  k_mix<<<dim3(4, 256, 8), 256, 0, stream>>>(x, y, mixedT);
  k_conv<<<dim3(4, 256, 8), 256, 0, stream>>>(mixedT, apack, conv_b, prelu_a,
                                              out, ssum, ssq);
  k_finstats<<<4, 256, 0, stream>>>(ssum, ssq, muv, rsg);
  k_attn<<<dim3(1024, 8), 256, 0, stream>>>(out, ppack, muv, rsg,
                                            keys_b, values_b, Sacc, Cacc);
  k_reproj<<<8, 128, 0, stream>>>(Sacc, Cacc, reproj_w, reproj_b, rvec);
  k_final<<<65536, 256, 0, stream>>>(out, muv, rsg, rvec);
}
